// LanguageModel_22084721836576
// MI455X (gfx1250) — matrix operand order repaired and checked by reading, not compiled, not run on hardware
//
#include <hip/hip_runtime.h>

// ---------------------------------------------------------------------------
// MI455X (gfx1250, wave32) implementation of the 12-layer hybrid LM forward.
// All dense matmuls (projections, MLP, QK^T, P*V) go through a single
// bf16 WMMA GEMM kernel (v_wmma_f32_16x16x32_bf16, fp32 accumulate) with
// double-buffered LDS staging via global_load_async_to_lds_b128 (ASYNCcnt).
// Block tile 128x64, per-wave 32x32 (4 WMMAs per K-step per wave).
// Two-pass attention exploits the 192MB L2 (scores are L2-resident).
// Gated delta rule uses LDS-resident 128x128 state (320KB LDS per WGP).
// ---------------------------------------------------------------------------

typedef unsigned short u16;
typedef __attribute__((ext_vector_type(16))) __bf16 v16bf;
typedef __attribute__((ext_vector_type(8)))  float  v8f;
typedef int v4i __attribute__((vector_size(16)));

#define AS1 __attribute__((address_space(1)))
#define AS3 __attribute__((address_space(3)))

#if __has_builtin(__builtin_amdgcn_global_load_async_to_lds_b128)
#define HAVE_ASYNC 1
static __device__ __forceinline__ void async_cp16(const u16* g, u16* l) {
  __builtin_amdgcn_global_load_async_to_lds_b128((AS1 v4i*)g, (AS3 v4i*)l, 0, 0);
}
#else
#define HAVE_ASYNC 0
#endif

#if __has_builtin(__builtin_amdgcn_s_wait_asynccnt)
#define WAIT_ASYNC() __builtin_amdgcn_s_wait_asynccnt(0)
#else
#define WAIT_ASYNC() asm volatile("s_wait_asynccnt 0" ::: "memory")
#endif

static __device__ __forceinline__ u16 f2bf(float f) {
  unsigned int u = __float_as_uint(f);
  u += 0x7fffu + ((u >> 16) & 1u);          // round-to-nearest-even
  return (u16)(u >> 16);
}
static __device__ __forceinline__ float sigmoidf_(float x) { return 1.f / (1.f + __expf(-x)); }

union Frag { v16bf v; uint4 q[2]; };

static __device__ __forceinline__ v8f wmma_bf16(v16bf a, v16bf b, v8f c) {
  return __builtin_amdgcn_wmma_f32_16x16x32_bf16(false, a, false, b, (short)0, c, false, false);
}

// ---------------------------------------------------------------------------
// Generic bf16 WMMA GEMM:  C[m,n] = sum_k A[m*lda+k] * W[n*ldw+k]
// Block: 256 threads (8 waves) -> tile 128(M) x 64(N), K-step 32.
// Wave grid 4(M) x 2(N); each wave computes 32x32 (4 accumulators, 4 WMMAs).
// Double-buffered LDS; async global->LDS copies (ASYNCcnt) when available.
// Requires: M % 128 == 0, K % 32 == 0 (true for all call sites), N arbitrary.
// ---------------------------------------------------------------------------
#define APITCH 40   // u16 elements per LDS row (80B -> 16B-aligned frags)

static __device__ __forceinline__ void stage_tiles(
    const u16* aSrc0, const u16* aSrc1, const u16* wSrc, bool nOK, int k0,
    u16* aDst0, u16* aDst1, u16* bDst)
{
#if HAVE_ASYNC
  async_cp16(aSrc0 + k0, aDst0);
  async_cp16(aSrc1 + k0, aDst1);
  if (nOK) {
    async_cp16(wSrc + k0, bDst);
  } else {
    uint4 z = make_uint4(0u, 0u, 0u, 0u);
    *(uint4*)bDst = z;                    // rows beyond N: zero via DS (no alias with async)
  }
#else
  *(uint4*)aDst0 = *(const uint4*)(aSrc0 + k0);
  *(uint4*)aDst1 = *(const uint4*)(aSrc1 + k0);
  uint4 d = make_uint4(0u, 0u, 0u, 0u);
  if (nOK) d = *(const uint4*)(wSrc + k0);
  *(uint4*)bDst = d;
#endif
}

__global__ __launch_bounds__(256)
void gemm_bf16_k(const u16* __restrict__ A, long lda,
                 const u16* __restrict__ W, long ldw,
                 float* __restrict__ C, long ldc,
                 int M, int N, int K)
{
  __shared__ __align__(16) u16 As[2][128 * APITCH];
  __shared__ __align__(16) u16 Bs[2][64 * APITCH];

  const int tid  = threadIdx.x;
  const int wave = tid >> 5;        // 0..7 (wave32)
  const int lane = tid & 31;
  const int wm   = wave & 3;        // M sub-tile (32 rows)
  const int wn   = wave >> 2;       // N sub-tile (32 cols)
  const int bm   = blockIdx.y * 128;
  const int bn   = blockIdx.x * 64;

  v8f acc00 = {}, acc01 = {}, acc10 = {}, acc11 = {};

  const int lrow = tid >> 2;        // 0..63
  const int lch  = tid & 3;         // 0..3  -> chunks of 8 bf16 (16B)
  const int half = lane >> 4;
  const int l16  = lane & 15;

  const u16* aSrc0 = A + (long)(bm + lrow) * lda + lch * 8;        // rows 0..63
  const u16* aSrc1 = aSrc0 + 64 * lda;                             // rows 64..127
  const int  nrow  = bn + lrow;
  const u16* wSrc  = W + (long)nrow * ldw + lch * 8;
  const bool nOK   = nrow < N;
  const int  soff0 = lrow * APITCH + lch * 8;
  const int  soff1 = (lrow + 64) * APITCH + lch * 8;

  // prologue: stage k0 = 0 into buffer 0
  stage_tiles(aSrc0, aSrc1, wSrc, nOK, 0, &As[0][soff0], &As[0][soff1], &Bs[0][soff0]);

  int buf = 0;
  for (int k0 = 0; k0 < K; k0 += 32, buf ^= 1) {
#if HAVE_ASYNC
    WAIT_ASYNC();
#endif
    __syncthreads();                // staged tiles visible; prior reads retired
    if (k0 + 32 < K)
      stage_tiles(aSrc0, aSrc1, wSrc, nOK, k0 + 32,
                  &As[buf ^ 1][soff0], &As[buf ^ 1][soff1], &Bs[buf ^ 1][soff0]);

    // ---- fragments (ISA 7.12.2 wave32 layouts) ----
    Frag a0, a1, b0, b1;
    {
      const u16* ar0 = &As[buf][(wm * 32 + l16) * APITCH];
      a0.q[0] = *(const uint4*)(ar0 + half * 8);        // K = half*8 .. +7
      a0.q[1] = *(const uint4*)(ar0 + 16 + half * 8);   // K = 16+half*8 .. +7
      const u16* ar1 = &As[buf][(wm * 32 + 16 + l16) * APITCH];
      a1.q[0] = *(const uint4*)(ar1 + half * 8);
      a1.q[1] = *(const uint4*)(ar1 + 16 + half * 8);
      const u16* br0 = &Bs[buf][(wn * 32 + l16) * APITCH + half * 8];
      b0.q[0] = *(const uint4*)(br0);
      b0.q[1] = *(const uint4*)(br0 + 16);
      const u16* br1 = &Bs[buf][(wn * 32 + 16 + l16) * APITCH + half * 8];
      b1.q[0] = *(const uint4*)(br1);
      b1.q[1] = *(const uint4*)(br1 + 16);
    }
    acc00 = wmma_bf16(a0.v, b0.v, acc00);
    acc01 = wmma_bf16(a0.v, b1.v, acc01);
    acc10 = wmma_bf16(a1.v, b0.v, acc10);
    acc11 = wmma_bf16(a1.v, b1.v, acc11);
  }

  // ---- epilogue: C/D layout M = r + 8*half, N = l16 ----
#pragma unroll
  for (int r = 0; r < 8; ++r) {
    int m0 = bm + wm * 32 + r + half * 8;      // a0 rows
    int m1 = m0 + 16;                          // a1 rows
    int n0 = bn + wn * 32 + l16;
    int n1 = n0 + 16;
    if (n0 < N) { C[(long)m0 * ldc + n0] = acc00[r]; C[(long)m1 * ldc + n0] = acc10[r]; }
    if (n1 < N) { C[(long)m0 * ldc + n1] = acc01[r]; C[(long)m1 * ldc + n1] = acc11[r]; }
  }
}

// ---------------------------------------------------------------------------
// Transpose + f32->bf16 convert:  out[c*R + r] = bf16(in[r*Cc + c])
// (used to pre-transpose V so P@V uses the fast GEMM path)
// ---------------------------------------------------------------------------
__global__ __launch_bounds__(256)
void transpose_cvt_k(const float* __restrict__ in, u16* __restrict__ out,
                     int R, int Cc)
{
  __shared__ float tile[32][33];
  int rb = blockIdx.y * 32, cb = blockIdx.x * 32;
  int tx = threadIdx.x & 31, ty = threadIdx.x >> 5;   // 32 x 8
  for (int r = ty; r < 32; r += 8) tile[r][tx] = in[(long)(rb + r) * Cc + cb + tx];
  __syncthreads();
  for (int r = ty; r < 32; r += 8) out[(long)(cb + r) * R + rb + tx] = f2bf(tile[tx][r]);
}

// ---------------------------------------------------------------------------
// Elementwise / normalization kernels
// ---------------------------------------------------------------------------
__global__ __launch_bounds__(256)
void rmsnorm_k(const float* __restrict__ x, const float* __restrict__ w,
               float* __restrict__ y, int width)  // scale = (1 + w)
{
  int row = blockIdx.x;
  const float* xr = x + (long)row * width;
  __shared__ float red[256];
  float s = 0.f;
  for (int i = threadIdx.x; i < width; i += 256) { float v = xr[i]; s += v * v; }
  red[threadIdx.x] = s; __syncthreads();
  for (int o = 128; o > 0; o >>= 1) { if (threadIdx.x < o) red[threadIdx.x] += red[threadIdx.x + o]; __syncthreads(); }
  float inv = rsqrtf(red[0] / (float)width + 1e-6f);
  for (int i = threadIdx.x; i < width; i += 256)
    y[(long)row * width + i] = xr[i] * inv * (1.f + w[i]);
}

__global__ __launch_bounds__(256)
void cvt_bf16_k(const float* __restrict__ x, u16* __restrict__ y, long n)
{
  long i = (long)blockIdx.x * 256 + threadIdx.x;
  if (i < n) y[i] = f2bf(x[i]);
}

__global__ __launch_bounds__(256)
void add_k(float* __restrict__ dst, const float* __restrict__ a,
           const float* __restrict__ b, long n)
{
  long i = (long)blockIdx.x * 256 + threadIdx.x;
  if (i < n) dst[i] = a[i] + b[i];
}

__global__ __launch_bounds__(256)
void silu_mul_bf16_k(const float* __restrict__ g, const float* __restrict__ u,
                     u16* __restrict__ out, long n)
{
  long i = (long)blockIdx.x * 256 + threadIdx.x;
  if (i < n) { float gv = g[i]; out[i] = f2bf(gv * sigmoidf_(gv) * u[i]); }
}

// rms_norm(head-dim 256, (1+w)) + partial RoPE (first 64 dims) -> bf16
__global__ __launch_bounds__(256)
void qk_norm_rope_k(const float* __restrict__ src, long outer, long inner, int nh,
                    const float* __restrict__ w, const float* __restrict__ cosb,
                    const float* __restrict__ sinb, u16* __restrict__ dst)
{
  int b = blockIdx.x;
  int t = b / nh, h = b % nh;
  const float* xr = src + (long)t * outer + (long)h * inner;
  __shared__ float red[256];
  __shared__ float xv[256];
  int i = threadIdx.x;                   // exactly 256 threads == head dim
  float v = xr[i];
  red[i] = v * v; __syncthreads();
  for (int o = 128; o > 0; o >>= 1) { if (i < o) red[i] += red[i + o]; __syncthreads(); }
  float inv = rsqrtf(red[0] / 256.f + 1e-6f);
  float q = v * inv * (1.f + w[i]);
  xv[i] = q; __syncthreads();
  float outv = q;
  if (i < 64) {
    float c = cosb[(long)t * 64 + i], s = sinb[(long)t * 64 + i];
    float rot = (i < 32) ? -xv[i + 32] : xv[i - 32];
    outv = q * c + rot * s;
  }
  dst[((long)t * nh + h) * 256 + i] = f2bf(outv);
}

// causal softmax over row (scale applied), emit bf16 probabilities
__global__ __launch_bounds__(256)
void softmax_causal_k(const float* __restrict__ S, u16* __restrict__ P,
                      int T, float scale)
{
  int row = blockIdx.x;
  const float* sr = S + (long)row * T;
  __shared__ float red[256];
  int tid = threadIdx.x;
  float mx = -1e30f;
  for (int j = tid; j <= row; j += 256) mx = fmaxf(mx, sr[j] * scale);
  red[tid] = mx; __syncthreads();
  for (int o = 128; o > 0; o >>= 1) { if (tid < o) red[tid] = fmaxf(red[tid], red[tid + o]); __syncthreads(); }
  mx = red[0]; __syncthreads();
  float sm = 0.f;
  for (int j = tid; j <= row; j += 256) sm += __expf(sr[j] * scale - mx);
  red[tid] = sm; __syncthreads();
  for (int o = 128; o > 0; o >>= 1) { if (tid < o) red[tid] += red[tid + o]; __syncthreads(); }
  float inv = 1.f / red[0];
  for (int j = tid; j < T; j += 256) {
    float p = (j <= row) ? __expf(sr[j] * scale - mx) * inv : 0.f;
    P[(long)row * T + j] = f2bf(p);
  }
}

// attn_out * sigmoid(gate-half of qg) -> bf16
__global__ __launch_bounds__(256)
void gate_out_k(const float* __restrict__ attn, const float* __restrict__ qg,
                u16* __restrict__ out)
{
  long idx = (long)blockIdx.x * 256 + threadIdx.x;   // T*2048
  int t = (int)(idx >> 11);
  int c = (int)(idx & 2047);
  int h = c >> 8, d = c & 255;
  float g = qg[(long)t * 4096 + h * 512 + 256 + d];
  out[idx] = f2bf(attn[idx] * sigmoidf_(g));
}

// causal depthwise conv1d (K=4) + SiLU
__global__ __launch_bounds__(256)
void conv_silu_k(const float* __restrict__ in, const float* __restrict__ w,
                 float* __restrict__ out, int T, int C)
{
  long idx = (long)blockIdx.x * 256 + threadIdx.x;
  if (idx >= (long)T * C) return;
  int c = (int)(idx % C);
  int t = (int)(idx / C);
  const float* wc = w + c * 4;
  float acc = 0.f;
#pragma unroll
  for (int j = 0; j < 4; ++j) {
    int tt = t - 3 + j;
    if (tt >= 0) acc += wc[j] * in[(long)tt * C + c];
  }
  out[idx] = acc * sigmoidf_(acc);
}

// beta = sigmoid(blin); g = -exp(A_log[h]) * softplus(alin + dt_bias[h])
__global__ __launch_bounds__(256)
void beta_g_k(const float* __restrict__ bl, const float* __restrict__ al,
              const float* __restrict__ alog, const float* __restrict__ dtb,
              float* __restrict__ beta, float* __restrict__ g, int n)
{
  int idx = blockIdx.x * 256 + threadIdx.x;
  if (idx >= n) return;
  int h = idx & 15;
  beta[idx] = sigmoidf_(bl[idx]);
  float a = al[idx] + dtb[h];
  float sp = (a > 20.f) ? a : log1pf(__expf(a));
  g[idx] = -__expf(alog[h]) * sp;
}

// l2norm(q)*Dk^-0.5 and l2norm(k) from conv output (q at 0, k at 2048)
__global__ __launch_bounds__(128)
void l2norm_qk_k(const float* __restrict__ conv, float* __restrict__ qn,
                 float* __restrict__ kn)
{
  int b = blockIdx.x;
  int t = b >> 4, h = b & 15, i = threadIdx.x;
  const float* q = conv + (long)t * 6144 + h * 128;
  const float* k = conv + (long)t * 6144 + 2048 + h * 128;
  __shared__ float r1[128], r2[128];
  float qv = q[i], kv = k[i];
  r1[i] = qv * qv; r2[i] = kv * kv; __syncthreads();
  for (int o = 64; o > 0; o >>= 1) { if (i < o) { r1[i] += r1[i + o]; r2[i] += r2[i + o]; } __syncthreads(); }
  float qi = rsqrtf(r1[0] + 1e-6f) * 0.08838834764831845f;  // * 128^-0.5
  float ki = rsqrtf(r2[0] + 1e-6f);
  long o2 = (long)t * 2048 + h * 128 + i;
  qn[o2] = qv * qi; kn[o2] = kv * ki;
}

// per-(head, chunk) inclusive cumsum of g over the 64-long chunk
__global__ __launch_bounds__(64)
void dr_cumsum_k(const float* __restrict__ g, float* __restrict__ gc)
{
  int h = blockIdx.x & 15, c = blockIdx.x >> 4, i = threadIdx.x;
  __shared__ float sg[64];
  sg[i] = g[(long)(c * 64 + i) * 16 + h]; __syncthreads();
  for (int o = 1; o < 64; o <<= 1) {
    float add = (i >= o) ? sg[i - o] : 0.f;
    __syncthreads();
    sg[i] += add;
    __syncthreads();
  }
  gc[(long)(c * 64 + i) * 16 + h] = sg[i];
}

// per-(head, chunk): A, unit-lower-triangular solve T=(I-A)^-1, v_t, k_cumdecay
__global__ __launch_bounds__(256)
void dr_chunk_k(const float* __restrict__ knG, const float* __restrict__ conv,
                const float* __restrict__ gc, const float* __restrict__ beta,
                float* __restrict__ vtG, float* __restrict__ kcdG)
{
  int h = blockIdx.x & 15, c = blockIdx.x >> 4;
  __shared__ float ks[64][128];   // kn rows, later v_beta
  __shared__ float kb[64][128];   // k_beta, later k_beta*exp(g)
  __shared__ float Am[64][64];
  __shared__ float Tm[64][64];
  __shared__ float gcs[64], bts[64], egs[64];
  int tid = threadIdx.x;
  long t0 = (long)c * 64;

  if (tid < 64) {
    float gv = gc[(t0 + tid) * 16 + h];
    gcs[tid] = gv; egs[tid] = __expf(gv);
    bts[tid] = beta[(t0 + tid) * 16 + h];
  }
  __syncthreads();

  for (int r = tid; r < 64 * 128; r += 256) {
    int i = r >> 7, d = r & 127;
    float kv = knG[(t0 + i) * 2048 + h * 128 + d];
    ks[i][d] = kv; kb[i][d] = kv * bts[i];
  }
  __syncthreads();

  for (int r = tid; r < 64 * 64; r += 256) {
    int i = r >> 6, j = r & 63;
    float a = 0.f;
    if (i > j) {
      float dot = 0.f;
      for (int d = 0; d < 128; ++d) dot += kb[i][d] * ks[j][d];
      a = -dot * __expf(gcs[i] - gcs[j]);
    }
    Am[i][j] = a;
    Tm[i][j] = (i == j) ? 1.f : 0.f;
  }
  __syncthreads();

  // forward substitution: T[i][:] = I[i][:] + sum_{j<i} A[i][j] * T[j][:]
  for (int i = 1; i < 64; ++i) {
    float acc = 0.f;
    if (tid < 64) for (int j = 0; j < i; ++j) acc += Am[i][j] * Tm[j][tid];
    __syncthreads();
    if (tid < 64) Tm[i][tid] += acc;
    __syncthreads();
  }

  // replace ks <- v_beta, kb <- k_beta * exp(g)
  for (int r = tid; r < 64 * 128; r += 256) {
    int i = r >> 7, d = r & 127;
    ks[i][d] = conv[(t0 + i) * 6144 + 4096 + h * 128 + d] * bts[i];
    kb[i][d] *= egs[i];
  }
  __syncthreads();

  for (int r = tid; r < 64 * 128; r += 256) {
    int i = r >> 7, d = r & 127;
    float sv = 0.f, sk = 0.f;
    for (int j = 0; j <= i; ++j) {          // T is unit lower triangular
      float tv = Tm[i][j];
      sv += tv * ks[j][d];
      sk += tv * kb[j][d];
    }
    vtG [(t0 + i) * 2048 + h * 128 + d] = sv;
    kcdG[(t0 + i) * 2048 + h * 128 + d] = sk;
  }
}

// per-head sequential scan over 32 chunks; 128x128 fp32 state in LDS
__global__ __launch_bounds__(256)
void dr_scan_k(const float* __restrict__ qn, const float* __restrict__ kn,
               const float* __restrict__ vt, const float* __restrict__ kcd,
               const float* __restrict__ gc, float* __restrict__ core)
{
  int h = blockIdx.x;
  __shared__ float S[128][129];     // padded state (66KB of the 320KB WGP LDS)
  __shared__ float vnew[64][128];
  __shared__ float at[64][64];
  __shared__ float sg[64], eg[64], ed[64];
  int tid = threadIdx.x;

  for (int r = tid; r < 128 * 128; r += 256) S[r >> 7][r & 127] = 0.f;
  __syncthreads();

  for (int c = 0; c < 32; ++c) {
    long t0 = (long)c * 64;
    if (tid < 64) {
      float g = gc[(t0 + tid) * 16 + h];
      sg[tid] = g; eg[tid] = __expf(g);
    }
    __syncthreads();
    float gl = sg[63];

    // v_new = v_t - kcd @ S
    for (int r = tid; r < 64 * 128; r += 256) {
      int i = r >> 7, d = r & 127;
      const float* kr = kcd + (t0 + i) * 2048 + h * 128;
      float acc = vt[(t0 + i) * 2048 + h * 128 + d];
      for (int k2 = 0; k2 < 128; ++k2) acc -= kr[k2] * S[k2][d];
      vnew[i][d] = acc;
    }
    // attn = tril(q k^T * decay)
    for (int r = tid; r < 64 * 64; r += 256) {
      int i = r >> 6, j = r & 63;
      float a = 0.f;
      if (i >= j) {
        const float* qi = qn + (t0 + i) * 2048 + h * 128;
        const float* kj = kn + (t0 + j) * 2048 + h * 128;
        float dot = 0.f;
        for (int d = 0; d < 128; ++d) dot += qi[d] * kj[d];
        a = dot * __expf(sg[i] - sg[j]);
      }
      at[i][j] = a;
    }
    __syncthreads();

    // out = (q*exp(g)) @ S + attn @ v_new
    for (int r = tid; r < 64 * 128; r += 256) {
      int i = r >> 7, d = r & 127;
      const float* qi = qn + (t0 + i) * 2048 + h * 128;
      float acc = 0.f;
      for (int k2 = 0; k2 < 128; ++k2) acc += qi[k2] * S[k2][d];
      acc *= eg[i];
      for (int j = 0; j <= i; ++j) acc += at[i][j] * vnew[j][d];
      core[(t0 + i) * 2048 + h * 128 + d] = acc;
    }
    if (tid < 64) ed[tid] = __expf(gl - sg[tid]);
    __syncthreads();

    // S = S*exp(g_last) + (k*exp(g_last-g))^T @ v_new
    float egl = __expf(gl);
    for (int r = tid; r < 128 * 128; r += 256) {
      int k2 = r >> 7, d = r & 127;
      float acc = S[k2][d] * egl;
      for (int i = 0; i < 64; ++i)
        acc += kn[(t0 + i) * 2048 + h * 128 + k2] * ed[i] * vnew[i][d];
      S[k2][d] = acc;
    }
    __syncthreads();
  }
}

// gated rms norm: core * rsqrt(mean(core^2)+eps) * w * silu(z) -> bf16
__global__ __launch_bounds__(128)
void gated_norm_k(const float* __restrict__ core, const float* __restrict__ w,
                  const float* __restrict__ z, u16* __restrict__ dst)
{
  int b = blockIdx.x;
  int t = b >> 4, h = b & 15, i = threadIdx.x;
  long o = (long)t * 2048 + h * 128 + i;
  __shared__ float red[128];
  float cv = core[o];
  red[i] = cv * cv; __syncthreads();
  for (int s = 64; s > 0; s >>= 1) { if (i < s) red[i] += red[i + s]; __syncthreads(); }
  float inv = rsqrtf(red[0] / 128.f + 1e-6f);
  float zv = z[o];
  dst[o] = f2bf(cv * inv * w[i] * zv * sigmoidf_(zv));
}

// ---------------------------------------------------------------------------
// Host orchestration
// ---------------------------------------------------------------------------
static inline int cdiv(long a, long b) { return (int)((a + b - 1) / b); }

extern "C" void kernel_launch(void* const* d_in, const int* in_sizes, int n_in,
                              void* d_out, int out_size, void* d_ws, size_t ws_size,
                              hipStream_t stream)
{
  (void)in_sizes; (void)n_in; (void)out_size; (void)ws_size;
  constexpr int SEQ = 2048;

  const float* hidden = (const float*)d_in[0];
  const float* cosb   = (const float*)d_in[1];
  const float* sinb   = (const float*)d_in[2];
  // d_in[3] causal_mask, d_in[4] attention_mask: realized via j<=row in softmax

  char*  base = (char*)d_ws;
  size_t off  = 0;
  auto alloc = [&](size_t bytes) -> char* {
    char* p = base + off; off = (off + bytes + 255) & ~(size_t)255; return p;
  };
  float* hbuf  = (float*)alloc((size_t)SEQ * 1024 * 4);
  float* xbuf  = (float*)alloc((size_t)SEQ * 1024 * 4);
  float* t0    = (float*)alloc((size_t)SEQ * 6144 * 4);
  float* t1    = (float*)alloc((size_t)SEQ * 3584 * 4);
  float* t2    = (float*)alloc((size_t)SEQ * 6144 * 4);
  float* ybuf  = (float*)alloc((size_t)SEQ * 1024 * 4);
  u16*   abf   = (u16*)  alloc((size_t)SEQ * 6144 * 2);
  u16*   wbf   = (u16*)  alloc((size_t)6144 * 1024 * 2);
  float* qnb   = (float*)alloc((size_t)SEQ * 2048 * 4);  // lin-attn only
  float* knb   = (float*)alloc((size_t)SEQ * 2048 * 4);  // lin-attn only
  float* vtb   = (float*)alloc((size_t)SEQ * 2048 * 4);
  float* kcdb  = (float*)alloc((size_t)SEQ * 2048 * 4);
  float* corb  = (float*)alloc((size_t)SEQ * 2048 * 4);
  float* gcb   = (float*)alloc((size_t)SEQ * 16 * 4);
  float* betab = (float*)alloc((size_t)SEQ * 16 * 4);
  float* gbuf  = (float*)alloc((size_t)SEQ * 16 * 4);
  float* blin  = (float*)alloc((size_t)SEQ * 16 * 4);
  float* alin  = (float*)alloc((size_t)SEQ * 16 * 4);
  // full-attn overlays (temporally disjoint from lin-attn usage of qnb/knb)
  u16*   qbf      = (u16*)qnb;                          // SEQ*2048 bf16
  u16*   kbf      = (u16*)qnb + (size_t)SEQ * 2048;     // SEQ*512  bf16
  u16*   vtbf     = (u16*)qnb + (size_t)SEQ * 2048 + (size_t)SEQ * 512;  // 512 x SEQ bf16 (V^T)
  float* attn_out = knb;                                // SEQ*2048 f32

  auto gemm = [&](const u16* A, long lda, const u16* W, long ldw,
                  float* C, long ldc, int M, int N, int K) {
    dim3 g(cdiv(N, 64), M / 128);
    gemm_bf16_k<<<g, 256, 0, stream>>>(A, lda, W, ldw, C, ldc, M, N, K);
  };
  auto cvt = [&](const float* x, u16* y, long n) {
    cvt_bf16_k<<<cdiv(n, 256), 256, 0, stream>>>(x, y, n);
  };

  const float* hcur = hidden;
  int idx = 5;
  for (int layer = 0; layer < 12; ++layer) {
    bool full = (layer == 3 || layer == 7 || layer == 11);
    const float* input_ln = (const float*)d_in[idx + 0];
    const float* post_ln  = (const float*)d_in[idx + 1];
    const float* gate_w   = (const float*)d_in[idx + 2];
    const float* up_w     = (const float*)d_in[idx + 3];
    const float* down_w   = (const float*)d_in[idx + 4];

    // ---- attention block ----
    rmsnorm_k<<<SEQ, 256, 0, stream>>>(hcur, input_ln, xbuf, 1024);
    cvt(xbuf, abf, (long)SEQ * 1024);

    if (full) {
      const float* q_w  = (const float*)d_in[idx + 5];
      const float* k_w  = (const float*)d_in[idx + 6];
      const float* v_w  = (const float*)d_in[idx + 7];
      const float* o_w  = (const float*)d_in[idx + 8];
      const float* q_nw = (const float*)d_in[idx + 9];
      const float* k_nw = (const float*)d_in[idx + 10];
      idx += 11;

      cvt(q_w, wbf, 4096L * 1024); gemm(abf, 1024, wbf, 1024, t0, 4096, SEQ, 4096, 1024);
      cvt(k_w, wbf, 512L * 1024);  gemm(abf, 1024, wbf, 1024, t1, 512,  SEQ, 512,  1024);
      cvt(v_w, wbf, 512L * 1024);  gemm(abf, 1024, wbf, 1024, t2, 512,  SEQ, 512,  1024);

      qk_norm_rope_k<<<SEQ * 8, 256, 0, stream>>>(t0, 4096, 512, 8, q_nw, cosb, sinb, qbf);
      qk_norm_rope_k<<<SEQ * 2, 256, 0, stream>>>(t1, 512,  256, 2, k_nw, cosb, sinb, kbf);
      // V (SEQ x 512 f32) -> V^T (512 x SEQ bf16) so P@V uses the fast path
      transpose_cvt_k<<<dim3(512 / 32, SEQ / 32), 256, 0, stream>>>(t2, vtbf, SEQ, 512);

      for (int hh = 0; hh < 8; ++hh) {
        int kv = hh >> 2;
        // scores = q @ k^T   (L2-resident 16MB buffer)
        gemm(qbf + hh * 256, 2048, kbf + kv * 256, 512, t2, 2048, SEQ, SEQ, 256);
        softmax_causal_k<<<SEQ, 256, 0, stream>>>(t2, abf, SEQ, 0.0625f);
        // out_h = P @ V  with W = V^T rows (256 x SEQ)
        gemm(abf, 2048, vtbf + (long)kv * 256 * SEQ, SEQ,
             attn_out + hh * 256, 2048, SEQ, 256, SEQ);
      }
      gate_out_k<<<SEQ * 2048 / 256, 256, 0, stream>>>(attn_out, t0, abf);
      cvt(o_w, wbf, 1024L * 2048);
      gemm(abf, 2048, wbf, 2048, ybuf, 1024, SEQ, 1024, 2048);
    } else {
      const float* qkv_w  = (const float*)d_in[idx + 5];
      const float* z_w    = (const float*)d_in[idx + 6];
      const float* b_w    = (const float*)d_in[idx + 7];
      const float* a_w    = (const float*)d_in[idx + 8];
      const float* conv_w = (const float*)d_in[idx + 9];
      const float* dtb    = (const float*)d_in[idx + 10];
      const float* alog   = (const float*)d_in[idx + 11];
      const float* gnw    = (const float*)d_in[idx + 12];
      const float* out_w  = (const float*)d_in[idx + 13];
      idx += 14;

      cvt(qkv_w, wbf, 6144L * 1024); gemm(abf, 1024, wbf, 1024, t0, 6144, SEQ, 6144, 1024);
      cvt(z_w,   wbf, 2048L * 1024); gemm(abf, 1024, wbf, 1024, t1, 2048, SEQ, 2048, 1024);
      cvt(b_w,   wbf, 16L * 1024);   gemm(abf, 1024, wbf, 1024, blin, 16, SEQ, 16, 1024);
      cvt(a_w,   wbf, 16L * 1024);   gemm(abf, 1024, wbf, 1024, alin, 16, SEQ, 16, 1024);

      conv_silu_k<<<SEQ * 6144 / 256, 256, 0, stream>>>(t0, conv_w, t2, SEQ, 6144);
      beta_g_k<<<SEQ * 16 / 256, 256, 0, stream>>>(blin, alin, alog, dtb, betab, gbuf, SEQ * 16);
      l2norm_qk_k<<<SEQ * 16, 128, 0, stream>>>(t2, qnb, knb);
      dr_cumsum_k<<<16 * 32, 64, 0, stream>>>(gbuf, gcb);
      dr_chunk_k<<<16 * 32, 256, 0, stream>>>(knb, t2, gcb, betab, vtb, kcdb);
      dr_scan_k<<<16, 256, 0, stream>>>(qnb, knb, vtb, kcdb, gcb, corb);
      gated_norm_k<<<SEQ * 16, 128, 0, stream>>>(corb, gnw, t1, abf);
      cvt(out_w, wbf, 1024L * 2048);
      gemm(abf, 2048, wbf, 2048, ybuf, 1024, SEQ, 1024, 2048);
    }
    add_k<<<SEQ * 1024 / 256, 256, 0, stream>>>(hbuf, hcur, ybuf, (long)SEQ * 1024);
    hcur = hbuf;

    // ---- MLP block ----
    rmsnorm_k<<<SEQ, 256, 0, stream>>>(hcur, post_ln, xbuf, 1024);
    cvt(xbuf, abf, (long)SEQ * 1024);
    cvt(gate_w, wbf, 3584L * 1024); gemm(abf, 1024, wbf, 1024, t0, 3584, SEQ, 3584, 1024);
    cvt(up_w,   wbf, 3584L * 1024); gemm(abf, 1024, wbf, 1024, t1, 3584, SEQ, 3584, 1024);
    silu_mul_bf16_k<<<SEQ * 3584 / 256, 256, 0, stream>>>(t0, t1, abf, (long)SEQ * 3584);
    cvt(down_w, wbf, 1024L * 3584);
    gemm(abf, 3584, wbf, 3584, ybuf, 1024, SEQ, 1024, 3584);

    float* dst = (layer == 11) ? (float*)d_out : hbuf;
    add_k<<<SEQ * 1024 / 256, 256, 0, stream>>>(dst, hcur, ybuf, (long)SEQ * 1024);
    hcur = hbuf;
  }
}
